// self_attention_layer_swin_26388279067293
// MI455X (gfx1250) — matrix operand order repaired and checked by reading, not compiled, not run on hardware
//
#include <hip/hip_runtime.h>
#include <hip/hip_bf16.h>

typedef __attribute__((ext_vector_type(16))) __bf16 v16bf;
typedef __attribute__((ext_vector_type(8)))  __bf16 v8bf;
typedef __attribute__((ext_vector_type(8)))  float  v8f;

#define B_     2
#define C_     256
#define OUT_   256
#define HD_    32
#define H_     8
#define N_     40962
#define MROWS  (B_ * N_)              /* 81924 */
#define MPAD   81952                  /* 32 * 2561 */
#define NTOP   2562
#define TOPW   16
#define NDOWN  2562
#define DOWNW  19
#define MWIN   (NTOP * TOPW + NDOWN * DOWNW)   /* 89670 */
#define XWROWS (MWIN + 1)
#define MLPH   512
#define SCALE_ 0.1767766952966369f

__device__ __forceinline__ __bf16 f2bf(float f) {
    union { float f; unsigned u; } in;
    in.f = f;
    unsigned u = in.u;
    unsigned r = u + 0x7FFFu + ((u >> 16) & 1u);   // round-to-nearest-even
    unsigned short h = (unsigned short)(r >> 16);
    union { unsigned short s; __bf16 b; } out;
    out.s = h;
    return out.b;
}

// ---------------------------------------------------------------------------
// 1. transpose x (B,C,N) f32 -> xt (MPAD rows x C) bf16, row = b*N+n
// ---------------------------------------------------------------------------
__global__ void transpose_x_kernel(const float* __restrict__ x, __bf16* __restrict__ xt) {
    __shared__ float tile[32][33];
    int n0 = blockIdx.x << 5, c0 = blockIdx.y << 5, b = blockIdx.z;
    int tx = threadIdx.x, ty = threadIdx.y;
    for (int i = ty; i < 32; i += 8) {
        int n = n0 + tx;
        tile[i][tx] = (n < N_) ? x[((size_t)b * C_ + (c0 + i)) * N_ + n] : 0.f;
    }
    __syncthreads();
    for (int i = ty; i < 32; i += 8) {
        int n = n0 + i;
        if (n < N_) xt[((size_t)b * N_ + n) * C_ + c0 + tx] = f2bf(tile[tx][i]);
    }
}

// ---------------------------------------------------------------------------
// 2. pack weight (K x Nc, row-major f32) into WMMA B-fragment bf16 layout:
//    frag f = nt*KT + kt, element lane*16+e : W[kt*32 + (e/8)*16 + (lane/16)*8 + e%8][nt*16 + lane%16]
// ---------------------------------------------------------------------------
__global__ void pack_w_kernel(const float* __restrict__ W, __bf16* __restrict__ Wp,
                              int K, int Nc) {
    int t = blockIdx.x * 256 + threadIdx.x;
    if (t >= K * Nc) return;
    int f = t >> 9;
    int r = t & 511;
    int lane = r >> 4, e = r & 15;
    int KT = K >> 5;
    int nt = f / KT, kt = f - nt * KT;
    int k = kt * 32 + ((e >> 3) << 4) + ((lane >> 4) << 3) + (e & 7);
    int n = (nt << 4) + (lane & 15);
    Wp[t] = f2bf(W[(size_t)k * Nc + n]);
}

// ---------------------------------------------------------------------------
// 3. bf16 WMMA GEMM: Y = A(MPAD x K) @ W(K x Nc) + bias [+ addend]
//    one wave -> 32 rows x 64 cols (8 WMMA accumulators).
//    k-loop manually unrolled by 2 with two named fragment buffers so loads
//    land directly in the alternate buffer (no rotation moves, partial waits).
// ---------------------------------------------------------------------------
__device__ __forceinline__ void load_frags(
    const __bf16* __restrict__ Arow0, const __bf16* __restrict__ Arow1,
    const __bf16* __restrict__ Wp, int nb, int KT, int kt, int lane,
    v16bf& a0, v16bf& a1, v16bf b[4]) {
    v8bf lo0 = *(const v8bf*)(Arow0 + (size_t)kt * 32);
    v8bf hi0 = *(const v8bf*)(Arow0 + (size_t)kt * 32 + 16);
    v8bf lo1 = *(const v8bf*)(Arow1 + (size_t)kt * 32);
    v8bf hi1 = *(const v8bf*)(Arow1 + (size_t)kt * 32 + 16);
#pragma unroll
    for (int i = 0; i < 8; ++i) {
        a0[i] = lo0[i]; a0[i + 8] = hi0[i];
        a1[i] = lo1[i]; a1[i + 8] = hi1[i];
    }
#pragma unroll
    for (int j = 0; j < 4; ++j) {
        size_t offW = (((size_t)(nb * 4 + j) * KT + kt) << 9) + ((size_t)lane << 4);
        b[j] = *(const v16bf*)(Wp + offW);
    }
}

__device__ __forceinline__ void do_wmma(const v16bf& a0, const v16bf& a1,
                                        const v16bf b[4], v8f acc[8]) {
#pragma unroll
    for (int j = 0; j < 4; ++j) {
        acc[j] = __builtin_amdgcn_wmma_f32_16x16x32_bf16(
            false, a0, false, b[j], (short)0, acc[j], false, false);
        acc[4 + j] = __builtin_amdgcn_wmma_f32_16x16x32_bf16(
            false, a1, false, b[j], (short)0, acc[4 + j], false, false);
    }
}

__global__ __launch_bounds__(256) void gemm_bf16_kernel(
    const __bf16* __restrict__ A, const __bf16* __restrict__ Wp,
    const float* __restrict__ bias, const float* __restrict__ addend,
    float* __restrict__ Yf, __bf16* __restrict__ Yb, int K, int Nc) {
    int lane = threadIdx.x & 31;
    int gw = blockIdx.x * 8 + (threadIdx.x >> 5);
    int numNB = Nc >> 6;
    int totalWaves = (MPAD >> 5) * numNB;
    if (gw >= totalWaves) return;
    int mt = gw / numNB, nb = gw - mt * numNB;
    int m0 = mt << 5;
    int KT = K >> 5;                 // always even here (8 or 16)
    const __bf16* Arow0 = A + (size_t)(m0 + (lane & 15)) * K + ((lane >> 4) << 3);
    const __bf16* Arow1 = Arow0 + (size_t)16 * K;
    v8f acc[8] = {};
    v16bf a0A, a1A, bA[4];
    v16bf a0B, a1B, bB[4];
    load_frags(Arow0, Arow1, Wp, nb, KT, 0, lane, a0A, a1A, bA);
    int halfIters = KT >> 1;
    for (int t = 0; t < halfIters - 1; ++t) {
        load_frags(Arow0, Arow1, Wp, nb, KT, 2 * t + 1, lane, a0B, a1B, bB);
        do_wmma(a0A, a1A, bA, acc);
        load_frags(Arow0, Arow1, Wp, nb, KT, 2 * t + 2, lane, a0A, a1A, bA);
        do_wmma(a0B, a1B, bB, acc);
    }
    load_frags(Arow0, Arow1, Wp, nb, KT, KT - 1, lane, a0B, a1B, bB);
    do_wmma(a0A, a1A, bA, acc);
    do_wmma(a0B, a1B, bB, acc);
    // epilogue
    int cl = lane & 15;
    int rb = m0 + ((lane >> 4) << 3);
    int n0 = nb << 6;
#pragma unroll
    for (int half = 0; half < 2; ++half) {
        int rbase = rb + half * 16;
#pragma unroll
        for (int j = 0; j < 4; ++j) {
            int col = n0 + j * 16 + cl;
            float bv = bias[col];
            v8f a = acc[half * 4 + j];
#pragma unroll
            for (int r = 0; r < 8; ++r) {
                size_t o = (size_t)(rbase + r) * Nc + col;
                float v = a[r] + bv;
                if (addend) v += addend[o];
                if (Yf) Yf[o] = v;
                if (Yb) Yb[o] = f2bf(v);
            }
        }
    }
}

// ---------------------------------------------------------------------------
// 4. window attention: one wave per (b,h,window).
//    Q/K/V staged into LDS via gfx1250 async global->LDS copies (ASYNCcnt).
// ---------------------------------------------------------------------------
template <int W>
__global__ __launch_bounds__(256) void attn_kernel(
    const float* __restrict__ yqkv, const int* __restrict__ widx,
    float* __restrict__ xw, int nWin, int winBase) {
    __shared__ float qs[8][W][33];
    __shared__ float ks[8][W][33];
    __shared__ float vs[8][W][33];
    int lane = threadIdx.x & 31;
    int wv = threadIdx.x >> 5;
    int gw = blockIdx.x * 8 + wv;
    int w = gw % nWin;
    int bh = gw / nWin;          // b*H + h
    int h = bh & 7;
    int b = bh >> 3;
    const int* iw = widx + (size_t)w * W;
    for (int j = 0; j < W; ++j) {
        int node = iw[j];
        const float* gq = yqkv + ((size_t)b * N_ + node) * 768 + h * HD_ + lane;
        unsigned lq = (unsigned)(uintptr_t)&qs[wv][j][lane];
        unsigned lk = (unsigned)(uintptr_t)&ks[wv][j][lane];
        unsigned lv = (unsigned)(uintptr_t)&vs[wv][j][lane];
        // q row (offset 0), k row (+256 f32 = +1024B), v row (+512 f32 = +2048B)
        asm volatile("global_load_async_to_lds_b32 %0, %1, off"
                     :: "v"(lq), "v"(gq) : "memory");
        asm volatile("global_load_async_to_lds_b32 %0, %1, off offset:1024"
                     :: "v"(lk), "v"(gq) : "memory");
        asm volatile("global_load_async_to_lds_b32 %0, %1, off offset:2048"
                     :: "v"(lv), "v"(gq) : "memory");
    }
    asm volatile("s_wait_asynccnt 0x0" ::: "memory");
    __syncthreads();
    if (lane < W) {
        float q[32];
#pragma unroll
        for (int c = 0; c < 32; ++c) q[c] = qs[wv][lane][c] * SCALE_;
        float s[W];
        float mx = -1e30f;
#pragma unroll
        for (int k = 0; k < W; ++k) {
            float acc = 0.f;
#pragma unroll
            for (int c = 0; c < 32; ++c) acc += q[c] * ks[wv][k][c];
            s[k] = acc;
            mx = fmaxf(mx, acc);
        }
        float sum = 0.f;
#pragma unroll
        for (int k = 0; k < W; ++k) { s[k] = __expf(s[k] - mx); sum += s[k]; }
        float inv = 1.f / sum;
        size_t orow = ((size_t)bh * XWROWS + winBase + (size_t)w * W + lane) * HD_;
#pragma unroll
        for (int c = 0; c < 32; ++c) {
            float o = 0.f;
#pragma unroll
            for (int k = 0; k < W; ++k) o += s[k] * vs[wv][k][c];
            xw[orow + c] = o * inv;
        }
    }
}

// zero the pad row (index MWIN) of xw for each (b,h): 512 threads
__global__ void zero_xw_pad_kernel(float* __restrict__ xw) {
    int t = threadIdx.x;
    int bh = t >> 5, c = t & 31;
    xw[((size_t)bh * XWROWS + MWIN) * HD_ + c] = 0.f;
}

// ---------------------------------------------------------------------------
// 5. gather/average via reverse_idx -> xf (bf16)
// ---------------------------------------------------------------------------
__global__ __launch_bounds__(256) void gather_kernel(
    const float* __restrict__ xw, const int* __restrict__ rev,
    const float* __restrict__ cnt_inv, __bf16* __restrict__ xf, int Kmax) {
    int row = blockIdx.x;           // b*N + n
    int b = row / N_;
    int n = row - b * N_;
    int t = threadIdx.x;
    int h = t >> 5, hd = t & 31;
    size_t base = (size_t)(b * H_ + h) * XWROWS;
    float acc = 0.f;
    for (int k = 0; k < Kmax; ++k) {
        int m = rev[(size_t)n * Kmax + k];
        acc += xw[(base + m) * HD_ + hd];
    }
    xf[(size_t)row * C_ + t] = f2bf(acc * cnt_inv[n]);
}

// ---------------------------------------------------------------------------
// 6. BN stats (sum, sumsq per channel) + finalize with transpose
// ---------------------------------------------------------------------------
__global__ void bn_zero_kernel(float* __restrict__ sums) {
    sums[blockIdx.x * 256 + threadIdx.x] = 0.f;
}

__global__ __launch_bounds__(256) void bn_stats_kernel(
    const float* __restrict__ res2, float* __restrict__ sums) {
    int t = threadIdx.x;
    int r0 = blockIdx.x << 8;
    int r1 = r0 + 256; if (r1 > MROWS) r1 = MROWS;
    float s = 0.f, s2 = 0.f;
    for (int r = r0; r < r1; ++r) {
        float v = res2[(size_t)r * OUT_ + t];
        s += v; s2 += v * v;
    }
    atomicAdd(&sums[t], s);
    atomicAdd(&sums[256 + t], s2);
}

__global__ __launch_bounds__(256) void final_kernel(
    const float* __restrict__ outf, const float* __restrict__ res2,
    const float* __restrict__ sums, const float* __restrict__ gamma,
    const float* __restrict__ beta, float* __restrict__ O) {
    __shared__ float tile[256][33];
    int ntiles = (N_ + 31) >> 5;
    int b = blockIdx.x / ntiles;
    int n0 = (blockIdx.x - b * ntiles) << 5;
    int t = threadIdx.x;
    float mean = sums[t] * (1.f / (float)MROWS);
    float var  = sums[256 + t] * (1.f / (float)MROWS) - mean * mean;
    float g  = rsqrtf(var + 1e-5f) * gamma[t];
    float bt = beta[t];
    for (int nn = 0; nn < 32; ++nn) {
        int n = n0 + nn;
        float v = 0.f;
        if (n < N_) {
            size_t o = ((size_t)b * N_ + n) * OUT_ + t;
            v = outf[o] + (res2[o] - mean) * g + bt;
        }
        tile[t][nn] = v;
    }
    __syncthreads();
    int wv = t >> 5, lane = t & 31;
    int n = n0 + lane;
    if (n < N_) {
#pragma unroll
        for (int j = 0; j < 32; ++j) {
            int c = wv + (j << 3);
            O[((size_t)b * OUT_ + c) * N_ + n] = tile[c][lane];
        }
    }
}

// ---------------------------------------------------------------------------
extern "C" void kernel_launch(void* const* d_in, const int* in_sizes, int n_in,
                              void* d_out, int out_size, void* d_ws, size_t ws_size,
                              hipStream_t stream) {
    (void)n_in; (void)out_size; (void)ws_size;
    const float* x      = (const float*)d_in[0];
    const float* Wqkv   = (const float*)d_in[1];
    const float* bqkv   = (const float*)d_in[2];
    const float* Wproj  = (const float*)d_in[3];
    const float* bproj  = (const float*)d_in[4];
    const float* Wres   = (const float*)d_in[5];
    const float* bres   = (const float*)d_in[6];
    const float* Wm1    = (const float*)d_in[7];
    const float* bm1    = (const float*)d_in[8];
    const float* Wm2    = (const float*)d_in[9];
    const float* bm2    = (const float*)d_in[10];
    const float* gamma  = (const float*)d_in[11];
    const float* beta   = (const float*)d_in[12];
    const float* cnt_inv = (const float*)d_in[13];
    const int* top_idx  = (const int*)d_in[14];
    const int* down_idx = (const int*)d_in[15];
    const int* rev      = (const int*)d_in[16];
    int Kmax = in_sizes[16] / N_;

    char* ws = (char*)d_ws;
    size_t off = 0;
    auto alloc = [&](size_t bytes) -> char* {
        char* p = ws + off;
        off = (off + bytes + 255) & ~(size_t)255;
        return p;
    };

    __bf16* xt   = (__bf16*)alloc((size_t)MPAD * C_ * 2);
    __bf16* wq_p = (__bf16*)alloc((size_t)C_ * 768 * 2);
    __bf16* wr_p = (__bf16*)alloc((size_t)C_ * OUT_ * 2);
    __bf16* wp_p = (__bf16*)alloc((size_t)C_ * OUT_ * 2);
    __bf16* w1_p = (__bf16*)alloc((size_t)OUT_ * MLPH * 2);
    __bf16* w2_p = (__bf16*)alloc((size_t)MLPH * OUT_ * 2);
    float*  yqkv = (float*)alloc((size_t)MPAD * 768 * 4);
    float*  resb = (float*)alloc((size_t)MPAD * OUT_ * 4);
    float*  xw   = (float*)alloc((size_t)B_ * H_ * XWROWS * HD_ * 4);
    __bf16* xf   = (__bf16*)alloc((size_t)MPAD * C_ * 2);
    float*  outf = (float*)alloc((size_t)MPAD * OUT_ * 4);
    float*  sums = (float*)alloc(512 * 4);
    // aliases into dead regions:
    __bf16* outb = xt;                                   // xt dead after GEMM1/2
    __bf16* hid  = (__bf16*)yqkv;                        // yqkv dead after attention
    float*  res2 = (float*)((char*)yqkv + (size_t)MPAD * MLPH * 2);

    // 1. transpose + bf16 convert
    {
        dim3 grid((N_ + 31) / 32, C_ / 32, B_);
        dim3 block(32, 8);
        transpose_x_kernel<<<grid, block, 0, stream>>>(x, xt);
    }
    // 2. pack weights
    auto packW = [&](const float* W, __bf16* Wp, int K, int Nc) {
        int tot = K * Nc;
        pack_w_kernel<<<(tot + 255) / 256, 256, 0, stream>>>(W, Wp, K, Nc);
    };
    packW(Wqkv, wq_p, C_, 768);
    packW(Wres, wr_p, C_, OUT_);
    packW(Wproj, wp_p, C_, OUT_);
    packW(Wm1, w1_p, OUT_, MLPH);
    packW(Wm2, w2_p, MLPH, OUT_);

    auto gemm = [&](const __bf16* A, const __bf16* Wp, const float* bias,
                    const float* add, float* Yf, __bf16* Yb, int K, int Nc) {
        int totalWaves = (MPAD / 32) * (Nc / 64);
        int blocks = (totalWaves + 7) / 8;
        gemm_bf16_kernel<<<blocks, 256, 0, stream>>>(A, Wp, bias, add, Yf, Yb, K, Nc);
    };

    // 3. QKV and residual GEMMs
    gemm(xt, wq_p, bqkv, nullptr, yqkv, nullptr, C_, 768);
    gemm(xt, wr_p, bres, nullptr, resb, nullptr, C_, OUT_);

    // 4. window attention
    attn_kernel<TOPW><<<(B_ * H_ * NTOP) / 8, 256, 0, stream>>>(yqkv, top_idx, xw, NTOP, 0);
    attn_kernel<DOWNW><<<(B_ * H_ * NDOWN) / 8, 256, 0, stream>>>(yqkv, down_idx, xw, NDOWN, NTOP * TOPW);
    zero_xw_pad_kernel<<<1, 512, 0, stream>>>(xw);

    // 5. gather + average -> xf (bf16)
    gather_kernel<<<MROWS, 256, 0, stream>>>(xw, rev, cnt_inv, xf, Kmax);

    // 6. proj GEMM (+res), producing fp32 out and bf16 copy for MLP
    gemm(xf, wp_p, bproj, resb, outf, outb, C_, OUT_);

    // 7. MLP GEMMs (hid/res2 overlay the dead yqkv buffer)
    gemm(outb, w1_p, bm1, nullptr, nullptr, hid, OUT_, MLPH);
    gemm(hid, w2_p, bm2, nullptr, res2, nullptr, MLPH, OUT_);

    // 8. BatchNorm + final transpose-add
    bn_zero_kernel<<<2, 256, 0, stream>>>(sums);
    bn_stats_kernel<<<(MROWS + 255) / 256, 256, 0, stream>>>(res2, sums);
    final_kernel<<<B_ * ((N_ + 31) / 32), 256, 0, stream>>>(
        outf, res2, sums, gamma, beta, (float*)d_out);
}
